// PyramidalNeuron_8358006358520
// MI455X (gfx1250) — compile-verified, matrix operand order checked by reading, not run on hardware
//
#include <hip/hip_runtime.h>
#include <hip/hip_bf16.h>

// ---------------------------------------------------------------------------
// PyramidalNeuron reduction for MI455X (gfx1250, wave32).
//
// out = f32(count(sensory > 0.5 || branches > 0)) - f32(count(basal > 0))
//
// Memory-bound: 768 MB read once -> ~33 us floor at 23.3 TB/s.
// Hot loop: 128-bit non-temporal global loads + integer compare/accumulate.
// Wave reduction: V_WMMA_F32_16X16X32_F16 used as a 32-lane sum engine
// (exact: per-lane counts <= 64 << 2048, the f16 exact-integer limit).
// Global accumulation is integer (counts ~33.5M exceed f32's 2^24 exact
// range); conversion to f32 happens per-count at the very end, matching
// the reference's f32(n_syn) - f32(n_feat) semantics.
// ---------------------------------------------------------------------------

typedef __attribute__((ext_vector_type(4)))  float    v4f;
typedef __attribute__((ext_vector_type(16))) _Float16 v16h;
typedef __attribute__((ext_vector_type(8)))  float    v8f;

#define COUNT_BLOCKS  4096
#define COUNT_THREADS 256

__global__ void pn_zero_kernel(int* __restrict__ acc) {
    acc[0] = 0;
    acc[1] = 0;
}

__global__ __launch_bounds__(COUNT_THREADS)
void pn_count_kernel(const float* __restrict__ sensory,
                     const float* __restrict__ basal,
                     const float* __restrict__ branches,
                     int* __restrict__ acc,
                     int n4) {
    const v4f* s4 = (const v4f*)sensory;
    const v4f* f4 = (const v4f*)basal;
    const v4f* b4 = (const v4f*)branches;

    const int tid    = blockIdx.x * blockDim.x + threadIdx.x;
    const int stride = gridDim.x * blockDim.x;

    int c_syn  = 0;   // count(sensory > 0.5 || branches > 0)
    int c_feat = 0;   // count(basal > 0)

    for (int i = tid; i < n4; i += stride) {
        v4f sv = __builtin_nontemporal_load(&s4[i]);
        v4f bv = __builtin_nontemporal_load(&b4[i]);
        v4f fv = __builtin_nontemporal_load(&f4[i]);
#pragma unroll
        for (int k = 0; k < 4; ++k) {
            c_syn  += ((sv[k] > 0.5f) || (bv[k] > 0.0f)) ? 1 : 0;
            c_feat += (fv[k] > 0.0f) ? 1 : 0;
        }
    }

    // ---- wave32 reduction via WMMA ----------------------------------------
    // A[m][k]: lane L(0-15) supplies row M=L (K in {0..7,16..23}); lane L+16
    // supplies row M=L (K in {8..15,24..31}). Every lane broadcasts its count
    // into all 16 of its A halves, and B = 1/16 everywhere, so:
    //   D[m][n] = (1/16) * (16*c_m + 16*c_{m+16}) = c_m + c_{m+16}
    // Lane n<16 holds D[0..7][n]; lane n>=16 holds D[8..15][n-16]. Summing a
    // lane's 8 D values and adding the shfl_xor(16) partner's sum yields the
    // full 32-lane count in every lane. All intermediates are exact integers
    // (per-lane count <= 64; f16 exact to 2048; f32 exact far beyond).
    _Float16 hs = (_Float16)c_syn;
    _Float16 hf = (_Float16)c_feat;
    v16h amat_s, amat_f, bmat;
#pragma unroll
    for (int i = 0; i < 16; ++i) {
        amat_s[i] = hs;
        amat_f[i] = hf;
        bmat[i]   = (_Float16)0.0625f;  // 1/16, exact power of two
    }
    v8f acc_s = {};
    v8f acc_f = {};
    acc_s = __builtin_amdgcn_wmma_f32_16x16x32_f16(
        false, amat_s, false, bmat, (short)0, acc_s, false, false);
    acc_f = __builtin_amdgcn_wmma_f32_16x16x32_f16(
        false, amat_f, false, bmat, (short)0, acc_f, false, false);

    float ssum = acc_s[0] + acc_s[1] + acc_s[2] + acc_s[3] +
                 acc_s[4] + acc_s[5] + acc_s[6] + acc_s[7];
    float fsum = acc_f[0] + acc_f[1] + acc_f[2] + acc_f[3] +
                 acc_f[4] + acc_f[5] + acc_f[6] + acc_f[7];
    ssum += __shfl_xor(ssum, 16, 32);
    fsum += __shfl_xor(fsum, 16, 32);

    // One pair of integer atomics per wave (gridDim*blockDim/32 waves total).
    if ((threadIdx.x & 31) == 0) {
        atomicAdd(&acc[0], (int)ssum);
        atomicAdd(&acc[1], (int)fsum);
    }
}

__global__ void pn_finalize_kernel(const int* __restrict__ acc,
                                   float* __restrict__ out) {
    // Match reference: f32(n_syn) - f32(n_feat) (each rounded separately).
    out[0] = (float)acc[0] - (float)acc[1];
}

extern "C" void kernel_launch(void* const* d_in, const int* in_sizes, int n_in,
                              void* d_out, int out_size, void* d_ws, size_t ws_size,
                              hipStream_t stream) {
    // setup_inputs() order: sensory_input, basal_features, branches_synapses
    const float* sensory  = (const float*)d_in[0];
    const float* basal    = (const float*)d_in[1];
    const float* branches = (const float*)d_in[2];
    float* out = (float*)d_out;
    int*   acc = (int*)d_ws;   // acc[0] = n_syn, acc[1] = n_feat

    const int n  = in_sizes[0];   // 8192*8192, divisible by 4
    const int n4 = n >> 2;

    pn_zero_kernel<<<1, 1, 0, stream>>>(acc);
    pn_count_kernel<<<COUNT_BLOCKS, COUNT_THREADS, 0, stream>>>(
        sensory, basal, branches, acc, n4);
    pn_finalize_kernel<<<1, 1, 0, stream>>>(acc, out);
}
